// CombinedQKV_Einsum_29377576305073
// MI455X (gfx1250) — matrix operand order repaired and checked by reading, not compiled, not run on hardware
//
#include <hip/hip_runtime.h>
#include <hip/hip_bf16.h>

typedef __attribute__((ext_vector_type(16))) __bf16 v16bf;
typedef __attribute__((ext_vector_type(8)))  __bf16 v8bf;
typedef __attribute__((ext_vector_type(8)))  float  v8f;
typedef __attribute__((ext_vector_type(4)))  float  v4f;

#define NDMODEL 1024
#define NHEADS  16
#define HEADDIM 64
#define NBATCH  2
#define SEQ     2048
#define MTOT    (NBATCH*SEQ)   // 4096 rows of x
#define NQKV    (3*NDMODEL)    // 3072

union ABFrag { v16bf v; v8bf h[2]; };

// ---------------------------------------------------------------------------
// Software-pipelined wave-level K-loop: one 16x64 tile (4 accumulators).
// A bf16 row-major [M,K]; B bf16 row-major [N,K] (= column-major KxN).
// K must be a multiple of 64. Steady state (load-after-use order):
//   wmma(set0/k) -> refill set0 from k+64 -> wmma(set1/k+32) -> refill set1
// so every load clause has one full WMMA group of latency cover, and both
// halves get cascaded partial s_wait_loadcnt instead of a hard 0-wait.
// ---------------------------------------------------------------------------
__device__ __forceinline__ void wmma_kloop_bf16(
    const __bf16* __restrict__ A, int lda,
    const __bf16* __restrict__ Bm, int ldb,
    int mBase, int nBase, int K, v8f c[4])
{
  const int lane = threadIdx.x & 31;
  const int kh   = lane >> 4;      // which K-half this lane holds
  const int li   = lane & 15;      // row (A) / column (B/D) index
  const __bf16* arow = A + (size_t)(mBase + li) * lda + kh * 16;
  const __bf16* bcol[4];
#pragma unroll
  for (int t = 0; t < 4; ++t)
    bcol[t] = Bm + (size_t)(nBase + t * 16 + li) * ldb + kh * 16;

  ABFrag a0, a1;
  v16bf  b0[4], b1[4];
  // prologue: fill both sets (chunks 0 and 32)
  a0.h[0] = *(const v8bf*)(arow);
  a0.h[1] = *(const v8bf*)(arow + 8);
#pragma unroll
  for (int t = 0; t < 4; ++t) b0[t] = *(const v16bf*)(bcol[t]);
  a1.h[0] = *(const v8bf*)(arow + 32);
  a1.h[1] = *(const v8bf*)(arow + 40);
#pragma unroll
  for (int t = 0; t < 4; ++t) b1[t] = *(const v16bf*)(bcol[t] + 32);

  for (int k0 = 0; k0 < K; k0 += 64) {
#pragma unroll
    for (int t = 0; t < 4; ++t)
      c[t] = __builtin_amdgcn_wmma_f32_16x16x32_bf16(
          false, a0.v, false, b0[t], (short)0, c[t], false, false);
    if (k0 + 64 < K) {               // refill set0 right after its consumers
      __builtin_prefetch(arow + k0 + 128, 0, 3);
      a0.h[0] = *(const v8bf*)(arow + k0 + 64);
      a0.h[1] = *(const v8bf*)(arow + k0 + 72);
#pragma unroll
      for (int t = 0; t < 4; ++t) b0[t] = *(const v16bf*)(bcol[t] + k0 + 64);
    }
#pragma unroll
    for (int t = 0; t < 4; ++t)
      c[t] = __builtin_amdgcn_wmma_f32_16x16x32_bf16(
          false, a1.v, false, b1[t], (short)0, c[t], false, false);
    if (k0 + 96 < K) {               // refill set1 right after its consumers
      a1.h[0] = *(const v8bf*)(arow + k0 + 96);
      a1.h[1] = *(const v8bf*)(arow + k0 + 104);
#pragma unroll
      for (int t = 0; t < 4; ++t) b1[t] = *(const v16bf*)(bcol[t] + k0 + 96);
    }
  }
}

// ---------------------------------------------------------------------------
// fp32 -> bf16 bulk convert (grid-stride)
// ---------------------------------------------------------------------------
__global__ void cvt_kernel(const float* __restrict__ src, __bf16* __restrict__ dst, int n)
{
  int i = blockIdx.x * blockDim.x + threadIdx.x;
  const int stride = gridDim.x * blockDim.x;
  for (; i < n; i += stride) dst[i] = (__bf16)src[i];
}

// ---------------------------------------------------------------------------
// K1: qkv = x @ qkv_w^T + b ; scatter to Q[b,h,s,d], K[b,h,s,d], V^T[b,h,d,s]
// ---------------------------------------------------------------------------
__global__ __launch_bounds__(256) void qkv_gemm_kernel(
    const __bf16* __restrict__ xbf, const __bf16* __restrict__ wbf,
    const float* __restrict__ bias,
    __bf16* __restrict__ qout, __bf16* __restrict__ kout, __bf16* __restrict__ vtout)
{
  const int wave  = threadIdx.x >> 5;
  const int mBase = blockIdx.y * 128 + wave * 16;
  const int nBase = blockIdx.x * 64;
  v8f c[4] = {};
  wmma_kloop_bf16(xbf, NDMODEL, wbf, NDMODEL, mBase, nBase, NDMODEL, c);
  const int lane = threadIdx.x & 31, kh = lane >> 4, li = lane & 15;
#pragma unroll
  for (int t = 0; t < 4; ++t) {
    const int n = nBase + t * 16 + li;
    const float bv = bias[n];
    const int which = n >> 10, rem = n & 1023, h = rem >> 6, d = rem & 63;
#pragma unroll
    for (int r = 0; r < 8; ++r) {
      const int m = mBase + kh * 8 + r;          // global row = b*SEQ + s
      const int b = m >> 11, s = m & 2047;
      const float val = c[t][r] + bv;
      const size_t bh = (size_t)(b * NHEADS + h);
      if (which == 0)      qout[(bh * SEQ + s) * HEADDIM + d]   = (__bf16)val;
      else if (which == 1) kout[(bh * SEQ + s) * HEADDIM + d]   = (__bf16)val;
      else                 vtout[(bh * HEADDIM + d) * SEQ + s]  = (__bf16)val;
    }
  }
}

// ---------------------------------------------------------------------------
// Pass 1: raw scores = (Q K^T)/8 masked, written ONCE; flash-style online
// per-row max m and sum-of-exp l maintained in registers (half-wave shfl
// reductions) and spilled to small stats buffers at the end.
// One wave owns 16 rows and sweeps all 2048 columns; Q fragments (K=64)
// are hoisted out of the column loop.
// ---------------------------------------------------------------------------
__global__ __launch_bounds__(256) void scores_stats_kernel(
    const __bf16* __restrict__ q, const __bf16* __restrict__ k,
    const int* __restrict__ mask, float* __restrict__ wts,
    float* __restrict__ rowM, float* __restrict__ rowL)
{
  const int bh = blockIdx.y;
  const int b  = bh >> 4;
  const int wave  = threadIdx.x >> 5;
  const int mBase = blockIdx.x * 128 + wave * 16;
  const int lane = threadIdx.x & 31, kh = lane >> 4, li = lane & 15;
  const float NEG_INF = -__builtin_inff();

  const __bf16* qp = q + (size_t)bh * SEQ * HEADDIM;
  const __bf16* kp = k + (size_t)bh * SEQ * HEADDIM;   // [t, d] row-major = [N,K]
  float* wrow = wts + (size_t)bh * SEQ * SEQ;

  // hoist the two A (Q) fragments: K = 64 -> k0 in {0, 32}
  const __bf16* arow = qp + (size_t)(mBase + li) * HEADDIM + kh * 16;
  ABFrag a0, a1;
  a0.h[0] = *(const v8bf*)(arow);
  a0.h[1] = *(const v8bf*)(arow + 8);
  a1.h[0] = *(const v8bf*)(arow + 32);
  a1.h[1] = *(const v8bf*)(arow + 40);

  float mrun[8], lrun[8];
#pragma unroll
  for (int r = 0; r < 8; ++r) { mrun[r] = NEG_INF; lrun[r] = 0.f; }

  for (int nBase = 0; nBase < SEQ; nBase += 64) {
    v8f c[4] = {};
#pragma unroll
    for (int t = 0; t < 4; ++t) {
      const __bf16* bptr = kp + (size_t)(nBase + t * 16 + li) * HEADDIM + kh * 16;
      v16bf b0 = *(const v16bf*)(bptr);
      v16bf b1 = *(const v16bf*)(bptr + 32);
      c[t] = __builtin_amdgcn_wmma_f32_16x16x32_bf16(
          false, a0.v, false, b0, (short)0, c[t], false, false);
      c[t] = __builtin_amdgcn_wmma_f32_16x16x32_bf16(
          false, a1.v, false, b1, (short)0, c[t], false, false);
    }
    // scale + mask, store raw, update running stats
    float dead[4];
#pragma unroll
    for (int t = 0; t < 4; ++t)
      dead[t] = (mask[b * SEQ + nBase + t * 16 + li] == 0) ? NEG_INF : 0.f;
#pragma unroll
    for (int t = 0; t < 4; ++t) {
      const int n = nBase + t * 16 + li;
#pragma unroll
      for (int r = 0; r < 8; ++r) {
        float val = c[t][r] * 0.125f + dead[t];     // -inf if masked
        c[t][r] = val;
        wrow[(size_t)(mBase + kh * 8 + r) * SEQ + n] = val;
      }
    }
#pragma unroll
    for (int r = 0; r < 8; ++r) {
      float tmax = fmaxf(fmaxf(c[0][r], c[1][r]), fmaxf(c[2][r], c[3][r]));
#pragma unroll
      for (int o = 8; o > 0; o >>= 1) tmax = fmaxf(tmax, __shfl_xor(tmax, o, 32));
      const float mnew = fmaxf(mrun[r], tmax);
      float ps = __expf(c[0][r] - mnew) + __expf(c[1][r] - mnew) +
                 __expf(c[2][r] - mnew) + __expf(c[3][r] - mnew);
#pragma unroll
      for (int o = 8; o > 0; o >>= 1) ps += __shfl_xor(ps, o, 32);
      if (mnew > NEG_INF)
        lrun[r] = lrun[r] * __expf(mrun[r] - mnew) + ps;
      mrun[r] = mnew;
    }
  }
  if (li == 0) {
#pragma unroll
    for (int r = 0; r < 8; ++r) {
      const int row = mBase + kh * 8 + r;
      rowM[(size_t)bh * SEQ + row] = mrun[r];
      rowL[(size_t)bh * SEQ + row] = lrun[r];
    }
  }
}

// ---------------------------------------------------------------------------
// Pass 2 (fused): read raw scores, apply exp(x-m)/l in registers, write the
// normalized probabilities (the required weights output) AND feed the same
// register values as bf16 A-fragments straight into the AV WMMAs.
// B = V^T bf16 [d, t]. Head outputs interleave into attnout [B,S,D] bf16.
// ---------------------------------------------------------------------------
__global__ __launch_bounds__(256) void norm_av_kernel(
    float* __restrict__ wts, const __bf16* __restrict__ vt,
    const float* __restrict__ rowM, const float* __restrict__ rowL,
    __bf16* __restrict__ attnout)
{
  const int bh = blockIdx.y;
  const int b = bh >> 4, h = bh & 15;
  const int wave  = threadIdx.x >> 5;
  const int mBase = blockIdx.x * 128 + wave * 16;
  const int lane = threadIdx.x & 31, kh = lane >> 4, li = lane & 15;

  const int arowIdx = mBase + li;                       // A-fragment row
  float* arow = wts + ((size_t)bh * SEQ + arowIdx) * SEQ;
  const float m   = rowM[(size_t)bh * SEQ + arowIdx];
  const float l   = rowL[(size_t)bh * SEQ + arowIdx];
  const float inv = (l > 0.f) ? (1.f / l) : 0.f;
  const __bf16* vp = vt + (size_t)bh * SEQ * HEADDIM;   // [d=64, t=2048]

  v8f c[4] = {};
  for (int t0 = 0; t0 < SEQ; t0 += 32) {
    float* p0 = arow + t0 + kh * 16;       // 8 consecutive t (lane's lower run)
    float* p1 = arow + t0 + kh * 16 + 8;   // 8 consecutive t (lane's upper run)
    v4f f0a = *(const v4f*)(p0),     f0b = *(const v4f*)(p0 + 4);
    v4f f1a = *(const v4f*)(p1),     f1b = *(const v4f*)(p1 + 4);
    ABFrag a;
#pragma unroll
    for (int i = 0; i < 4; ++i) {
      float e;
      e = __expf(f0a[i] - m); f0a[i] = (inv > 0.f) ? e * inv : 0.f;
      e = __expf(f0b[i] - m); f0b[i] = (inv > 0.f) ? e * inv : 0.f;
      e = __expf(f1a[i] - m); f1a[i] = (inv > 0.f) ? e * inv : 0.f;
      e = __expf(f1b[i] - m); f1b[i] = (inv > 0.f) ? e * inv : 0.f;
    }
    *(v4f*)(p0)     = f0a;  *(v4f*)(p0 + 4) = f0b;      // normalized weights out
    *(v4f*)(p1)     = f1a;  *(v4f*)(p1 + 4) = f1b;
#pragma unroll
    for (int i = 0; i < 4; ++i) {
      a.h[0][i]     = (__bf16)f0a[i];  a.h[0][i + 4] = (__bf16)f0b[i];
      a.h[1][i]     = (__bf16)f1a[i];  a.h[1][i + 4] = (__bf16)f1b[i];
    }
#pragma unroll
    for (int t = 0; t < 4; ++t) {
      const __bf16* bptr = vp + (size_t)(t * 16 + li) * SEQ + t0 + kh * 16;
      v16bf bb = *(const v16bf*)bptr;
      c[t] = __builtin_amdgcn_wmma_f32_16x16x32_bf16(
          false, a.v, false, bb, (short)0, c[t], false, false);
    }
  }
#pragma unroll
  for (int t = 0; t < 4; ++t) {
    const int d = t * 16 + li;
#pragma unroll
    for (int r = 0; r < 8; ++r) {
      const int s = mBase + kh * 8 + r;
      attnout[((size_t)(b * SEQ + s)) * NDMODEL + h * HEADDIM + d] = (__bf16)c[t][r];
    }
  }
}

// ---------------------------------------------------------------------------
// K5: out = attnout @ wo_w^T + wo_b  (fp32 result)
// ---------------------------------------------------------------------------
__global__ __launch_bounds__(256) void proj_gemm_kernel(
    const __bf16* __restrict__ a, const __bf16* __restrict__ wbf,
    const float* __restrict__ bias, float* __restrict__ out)
{
  const int wave  = threadIdx.x >> 5;
  const int mBase = blockIdx.y * 128 + wave * 16;
  const int nBase = blockIdx.x * 64;
  v8f c[4] = {};
  wmma_kloop_bf16(a, NDMODEL, wbf, NDMODEL, mBase, nBase, NDMODEL, c);
  const int lane = threadIdx.x & 31, kh = lane >> 4, li = lane & 15;
#pragma unroll
  for (int t = 0; t < 4; ++t) {
    const int n = nBase + t * 16 + li;
    const float bv = bias[n];
#pragma unroll
    for (int r = 0; r < 8; ++r) {
      const int m = mBase + kh * 8 + r;
      out[(size_t)m * NDMODEL + n] = c[t][r] + bv;
    }
  }
}

// ---------------------------------------------------------------------------
extern "C" void kernel_launch(void* const* d_in, const int* in_sizes, int n_in,
                              void* d_out, int out_size, void* d_ws, size_t ws_size,
                              hipStream_t stream)
{
  const float* x     = (const float*)d_in[0];
  const int*   mask  = (const int*)d_in[1];
  const float* qkv_w = (const float*)d_in[2];
  const float* qkv_b = (const float*)d_in[3];
  const float* wo_w  = (const float*)d_in[4];
  const float* wo_b  = (const float*)d_in[5];

  float* out = (float*)d_out;                       // [B,S,D] fp32
  float* wts = out + (size_t)MTOT * NDMODEL;        // [B,H,S,S] fp32 (output #2)

  // workspace carve-up (bf16 + stats), ~49 MB total
  __bf16* xbf    = (__bf16*)d_ws;                          // 4M elem
  __bf16* qkvwbf = xbf    + (size_t)MTOT   * NDMODEL;      // 3M
  __bf16* wobf   = qkvwbf + (size_t)NQKV   * NDMODEL;      // 1M
  __bf16* qb     = wobf   + (size_t)NDMODEL* NDMODEL;      // 4M  Q[b,h,s,d]
  __bf16* kb     = qb     + (size_t)MTOT   * NDMODEL;      // 4M  K[b,h,t,d]
  __bf16* vtb    = kb     + (size_t)MTOT   * NDMODEL;      // 4M  V^T[b,h,d,t]
  __bf16* attnb  = vtb    + (size_t)MTOT   * NDMODEL;      // 4M  attnout[B,S,D]
  float*  rowM   = (float*)(attnb + (size_t)MTOT * NDMODEL); // 64K rows
  float*  rowL   = rowM + (size_t)NBATCH * NHEADS * SEQ;

  cvt_kernel<<<1024, 256, 0, stream>>>(x,     xbf,    MTOT * NDMODEL);
  cvt_kernel<<<1024, 256, 0, stream>>>(qkv_w, qkvwbf, NQKV * NDMODEL);
  cvt_kernel<<<256,  256, 0, stream>>>(wo_w,  wobf,   NDMODEL * NDMODEL);

  qkv_gemm_kernel<<<dim3(NQKV / 64, MTOT / 128), 256, 0, stream>>>(
      xbf, qkvwbf, qkv_b, qb, kb, vtb);

  scores_stats_kernel<<<dim3(SEQ / 128, NBATCH * NHEADS), 256, 0, stream>>>(
      qb, kb, mask, wts, rowM, rowL);

  norm_av_kernel<<<dim3(SEQ / 128, NBATCH * NHEADS), 256, 0, stream>>>(
      wts, vtb, rowM, rowL, attnb);

  proj_gemm_kernel<<<dim3(NDMODEL / 64, MTOT / 128), 256, 0, stream>>>(
      attnb, wobf, wo_b, out);
}
